// GemmaAttention_22076131901974
// MI455X (gfx1250) — matrix operand order repaired and checked by reading, not compiled, not run on hardware
//
#include <hip/hip_runtime.h>

// ---------------------------------------------------------------------------
// Gemma attention forward for gfx1250 (MI455X), wave32 + v_wmma_f32_16x16x32_bf16
//   out[0] = attn_output (B,S,2048) f32
//   out[1] = attn_weights (B,8,2048,2048) f32
// Pipeline:
//   k0: convert X -> bf16; transpose+convert Wq/Wk/Wv/Wo -> bf16 [n][k]
//   k1: QKV projection (WMMA, TDM-staged A, 32x256 tiles, 2Mx2N) + RoPE
//       -> Qb[b,h,s,d], Kb[b,s,d], Vt[b,d,s] bf16
//   k2: fused  S=QK^T*scale+mask -> softmax -> attn_weights f32 (non-temporal),
//       P(bf16) staged in LDS -> PV (WMMA) -> AOb[b,s,h*256+d] bf16
//   k3: output projection (WMMA, TDM-staged A, 32x256 tiles, 2Mx2N) -> f32
// ---------------------------------------------------------------------------

typedef __attribute__((ext_vector_type(16))) __bf16 bf16x16;
typedef __attribute__((ext_vector_type(8)))  float  f32x8;
typedef __attribute__((ext_vector_type(4)))  unsigned int v4u;
typedef __attribute__((ext_vector_type(8)))  int v8i;
typedef __attribute__((ext_vector_type(4)))  int v4i;

struct __align__(16) U4 { unsigned int x, y, z, w; };
union Frag { U4 u[2]; bf16x16 v; };

static __device__ __forceinline__ unsigned short f2bf(float f) {
    union { float f; unsigned int u; } x; x.f = f;
    unsigned int r = (x.u + 0x7FFFu + ((x.u >> 16) & 1u)) >> 16;   // RNE
    return (unsigned short)r;
}

static __device__ __forceinline__ f32x8 wmma_bf16(bf16x16 a, bf16x16 b, f32x8 c) {
    // D = A(16x32 bf16) x B(32x16 bf16) + C(16x16 f32)
    return __builtin_amdgcn_wmma_f32_16x16x32_bf16(
        false, a, false, b, (short)0, c, false, false);
}

#if defined(__has_builtin)
#if __has_builtin(__builtin_amdgcn_tensor_load_to_lds) && \
    __has_builtin(__builtin_amdgcn_s_wait_tensorcnt)
#define HAVE_TDM 1
#endif
#endif

#ifdef HAVE_TDM
// 2-D TDM tile load: tile (tile_d0 x tile_d1) elements of 2 bytes from a
// row-major tensor (tens_d0 wide, stride0 elems between rows) into LDS with
// 16B pad per 128B row (cfg packs data_size/pad fields of D# group1 word0).
static __device__ __forceinline__ void tdm_load_2d(
    unsigned lds_addr, unsigned long long gaddr,
    unsigned tile_d0, unsigned tile_d1,
    unsigned tens_d0, unsigned tens_d1,
    unsigned stride0, unsigned cfg_w0) {
    v4u g0;
    g0.x = 1u;                                               // count=1 (user D#)
    g0.y = lds_addr;                                         // LDS byte address
    g0.z = (unsigned)(gaddr & 0xFFFFFFFFull);                // global_addr[31:0]
    g0.w = (unsigned)((gaddr >> 32) & 0x01FFFFFFull)         // global_addr[56:32]
         | (2u << 30);                                       // type=2 (image)
    v8i g1;
    g1[0] = (int)cfg_w0;                                     // mask/data_size/pad
    g1[1] = (int)((tens_d0 & 0xFFFFu) << 16);                // tensor_dim0 lo
    g1[2] = (int)(((tens_d0 >> 16) & 0xFFFFu)
                  | ((tens_d1 & 0xFFFFu) << 16));            // dim0 hi | dim1 lo
    g1[3] = (int)(((tens_d1 >> 16) & 0xFFFFu)
                  | ((tile_d0 & 0xFFFFu) << 16));            // dim1 hi | tile0
    g1[4] = (int)(tile_d1 & 0xFFFFu);                        // tile1 (tile2=0)
    g1[5] = (int)stride0;                                    // dim0_stride lo32
    g1[6] = 0;
    g1[7] = 0;
    v4i z4 = {0, 0, 0, 0};
#if defined(__clang_major__) && __clang_major__ >= 23
    v8i z8 = {0, 0, 0, 0, 0, 0, 0, 0};
    __builtin_amdgcn_tensor_load_to_lds(g0, g1, z4, z4, z8, 0);
#else
    __builtin_amdgcn_tensor_load_to_lds(g0, g1, z4, z4, 0);
#endif
}
#endif

// ---- problem constants -----------------------------------------------------
#define BB   2
#define SSQ  2048
#define HHN  8
#define DDH  256
#define HIDN 2048
#define BSR  (BB * SSQ)          /* 4096 rows */

// A-chunk staging geometry (shared by k1/k3)
#define ACH_ROWS   32
#define ACH_K      64
#define ACH_STRH   72                    /* halfs per LDS row: 128B + 16B pad */
#define ACH_BYTES  (ACH_ROWS * ACH_STRH * 2)   /* 4608 */
#define NCHUNK     (HIDN / ACH_K)        /* 32 */
// group1 word0: data_size=2B (code 1), pad_enable, pad_interval=32 DWORDs,
// pad_amount=4 DWORDs (16B pad per 128B row)
#define TDM_CFG    ((1u << 16) | (1u << 20) | (4u << 22) | (3u << 25))

// ---- workspace layout (bytes) ---------------------------------------------
static const size_t OFF_XB  = 0;
static const size_t SZ_XB   = (size_t)BSR * HIDN * 2;          // 16 MB
static const size_t OFF_WQT = OFF_XB  + SZ_XB;
static const size_t SZ_WQT  = (size_t)HIDN * HIDN * 2;         // 8 MB
static const size_t OFF_WKT = OFF_WQT + SZ_WQT;
static const size_t SZ_WKT  = (size_t)DDH * HIDN * 2;          // 1 MB
static const size_t OFF_WVT = OFF_WKT + SZ_WKT;
static const size_t SZ_WVT  = SZ_WKT;
static const size_t OFF_WOT = OFF_WVT + SZ_WVT;
static const size_t SZ_WOT  = (size_t)HIDN * HIDN * 2;         // 8 MB
static const size_t OFF_QB  = OFF_WOT + SZ_WOT;
static const size_t SZ_QB   = (size_t)BB * HHN * SSQ * DDH * 2; // 16 MB
static const size_t OFF_KB  = OFF_QB  + SZ_QB;
static const size_t SZ_KB   = (size_t)BB * SSQ * DDH * 2;      // 2 MB
static const size_t OFF_VT  = OFF_KB  + SZ_KB;
static const size_t SZ_VT   = SZ_KB;
static const size_t OFF_AOB = OFF_VT  + SZ_VT;                 // 16 MB

// ===========================================================================
// k0a: flat f32 -> bf16
__global__ __launch_bounds__(256) void cvt_kernel(const float* __restrict__ src,
                                                  unsigned short* __restrict__ dst,
                                                  int n) {
    int i = blockIdx.x * 256 + threadIdx.x;
    if (i < n) dst[i] = f2bf(src[i]);
}

// k0b: transpose+convert: src is [K][N] f32, dst is [N][K] bf16
__global__ __launch_bounds__(256) void cvt_t_kernel(const float* __restrict__ src,
                                                    unsigned short* __restrict__ dst,
                                                    int K, int N) {
    int i = blockIdx.x * 256 + threadIdx.x;
    if (i < K * N) {
        int n = i / K;
        int k = i - n * K;
        dst[i] = f2bf(src[(size_t)k * N + n]);
    }
}

// ===========================================================================
// k1: QKV projection + RoPE.  grid = (BSR/32, 10):
//   y = 0..7 -> Q head y ; y == 8 -> K ; y == 9 -> V
// WG tile 32x256; wave w -> cols [32w, 32w+32), both 16-row halves (2Mx2N).
// X A-chunks (32x64 bf16) staged into LDS by the TDM, double buffered.
__global__ __launch_bounds__(256) void qkv_kernel(
    const unsigned short* __restrict__ Xb,
    const unsigned short* __restrict__ Wqt,
    const unsigned short* __restrict__ Wkt,
    const unsigned short* __restrict__ Wvt,
    const int* __restrict__ pid,
    unsigned short* __restrict__ Qb,
    unsigned short* __restrict__ Kb,
    unsigned short* __restrict__ Vt) {
    __shared__ unsigned short Abuf[2][ACH_ROWS * ACH_STRH];
    __shared__ float tile[32][260];                 // raw f32 outputs (padded)

    const int z     = blockIdx.y;
    const int tid   = threadIdx.x;
    const int wave  = tid >> 5;
    const int lane  = tid & 31;
    const int lhalf = lane & 15;
    const int lhi   = lane >> 4;
    const int m0    = blockIdx.x * 32;
    const int n0    = wave * 32;

    const unsigned short* Wt =
        (z < 8) ? (Wqt + (size_t)z * DDH * HIDN) : ((z == 8) ? Wkt : Wvt);
    const unsigned short* brow0 = Wt + (size_t)(n0 + lhalf) * HIDN;
    const unsigned short* brow1 = Wt + (size_t)(n0 + 16 + lhalf) * HIDN;

    f32x8 c00 = {}, c01 = {}, c10 = {}, c11 = {};

#ifdef HAVE_TDM
    const unsigned lds_base = (unsigned)(size_t)(&Abuf[0][0]);
    const unsigned long long gbase =
        (unsigned long long)(size_t)(Xb + (size_t)m0 * HIDN);
    if (wave == 0) {
        tdm_load_2d(lds_base, gbase, ACH_K, ACH_ROWS, HIDN, BSR, HIDN, TDM_CFG);
        tdm_load_2d(lds_base + ACH_BYTES, gbase + 2 * ACH_K,
                    ACH_K, ACH_ROWS, HIDN, BSR, HIDN, TDM_CFG);
        __builtin_amdgcn_s_wait_tensorcnt(1);      // chunk 0 complete
    }
    __syncthreads();
    for (int c = 0; c < NCHUNK; ++c) {
        const unsigned short* abuf = &Abuf[c & 1][0];
        const int kg = c * ACH_K;
#pragma unroll
        for (int ks = 0; ks < 2; ++ks) {
            const int kc = ks * 32;
            Frag a0, a1, b0, b1;
            const unsigned short* ar0 = abuf + (size_t)lhalf * ACH_STRH;
            const unsigned short* ar1 = abuf + (size_t)(16 + lhalf) * ACH_STRH;
            a0.u[0] = *(const U4*)(ar0 + kc + 16 * lhi);
            a0.u[1] = *(const U4*)(ar0 + kc + 16 * lhi + 8);
            a1.u[0] = *(const U4*)(ar1 + kc + 16 * lhi);
            a1.u[1] = *(const U4*)(ar1 + kc + 16 * lhi + 8);
            const int k0 = kg + kc;
            b0.u[0] = *(const U4*)(brow0 + k0 + 16 * lhi);
            b0.u[1] = *(const U4*)(brow0 + k0 + 16 * lhi + 8);
            b1.u[0] = *(const U4*)(brow1 + k0 + 16 * lhi);
            b1.u[1] = *(const U4*)(brow1 + k0 + 16 * lhi + 8);
            c00 = wmma_bf16(a0.v, b0.v, c00);
            c01 = wmma_bf16(a0.v, b1.v, c01);
            c10 = wmma_bf16(a1.v, b0.v, c10);
            c11 = wmma_bf16(a1.v, b1.v, c11);
        }
        __syncthreads();                    // all waves done reading buf[c&1]
        if (wave == 0) {
            if (c + 2 < NCHUNK) {
                tdm_load_2d(lds_base + (unsigned)((c & 1) * ACH_BYTES),
                            gbase + (unsigned long long)(c + 2) * (2 * ACH_K),
                            ACH_K, ACH_ROWS, HIDN, BSR, HIDN, TDM_CFG);
                __builtin_amdgcn_s_wait_tensorcnt(1);   // chunk c+1 complete
            } else {
                __builtin_amdgcn_s_wait_tensorcnt(0);   // drain
            }
        }
        __syncthreads();                    // publish chunk c+1 to all waves
    }
#else
    for (int c = 0; c < NCHUNK; ++c) {
        const int kg = c * ACH_K;
        __syncthreads();
        for (int i = tid; i < ACH_ROWS * ACH_K; i += 256) {
            int r = i >> 6, kk = i & 63;
            Abuf[0][r * ACH_STRH + kk] = Xb[(size_t)(m0 + r) * HIDN + kg + kk];
        }
        __syncthreads();
        const unsigned short* abuf = &Abuf[0][0];
#pragma unroll
        for (int ks = 0; ks < 2; ++ks) {
            const int kc = ks * 32;
            Frag a0, a1, b0, b1;
            const unsigned short* ar0 = abuf + (size_t)lhalf * ACH_STRH;
            const unsigned short* ar1 = abuf + (size_t)(16 + lhalf) * ACH_STRH;
            a0.u[0] = *(const U4*)(ar0 + kc + 16 * lhi);
            a0.u[1] = *(const U4*)(ar0 + kc + 16 * lhi + 8);
            a1.u[0] = *(const U4*)(ar1 + kc + 16 * lhi);
            a1.u[1] = *(const U4*)(ar1 + kc + 16 * lhi + 8);
            const int k0 = kg + kc;
            b0.u[0] = *(const U4*)(brow0 + k0 + 16 * lhi);
            b0.u[1] = *(const U4*)(brow0 + k0 + 16 * lhi + 8);
            b1.u[0] = *(const U4*)(brow1 + k0 + 16 * lhi);
            b1.u[1] = *(const U4*)(brow1 + k0 + 16 * lhi + 8);
            c00 = wmma_bf16(a0.v, b0.v, c00);
            c01 = wmma_bf16(a0.v, b1.v, c01);
            c10 = wmma_bf16(a1.v, b0.v, c10);
            c11 = wmma_bf16(a1.v, b1.v, c11);
        }
    }
#endif

    // park raw f32 results in LDS, then RoPE + bf16 store
#pragma unroll
    for (int r = 0; r < 8; ++r) {
        int row = r + 8 * lhi;
        tile[row][n0 + lhalf]           = c00[r];
        tile[row][n0 + 16 + lhalf]      = c01[r];
        tile[16 + row][n0 + lhalf]      = c10[r];
        tile[16 + row][n0 + 16 + lhalf] = c11[r];
    }
    __syncthreads();

    const int bi = m0 >> 11;            // batch
    const int s0 = m0 & (SSQ - 1);      // seq base
#pragma unroll
    for (int i = tid; i < 32 * DDH; i += 256) {
        int row = i >> 8, d = i & 255;
        int s = s0 + row;
        float val = tile[row][d];
        if (z < 9) {                               // RoPE for Q and K
            int pos = pid[bi * SSQ + s];
            float inv = __expf(-(float)(d & 127) * (9.210340371976184f / 128.0f));
            float ang = (float)pos * inv;
            float cs = __cosf(ang), sn = __sinf(ang);
            float other = (d < 128) ? tile[row][d + 128] : tile[row][d - 128];
            val = val * cs + ((d < 128) ? -other : other) * sn;
        }
        unsigned short hv = f2bf(val);
        if (z < 8)
            Qb[(((size_t)bi * HHN + z) * SSQ + s) * DDH + d] = hv;
        else if (z == 8)
            Kb[((size_t)bi * SSQ + s) * DDH + d] = hv;
        else
            Vt[((size_t)bi * DDH + d) * SSQ + s] = hv;   // store V transposed
    }
}

// ===========================================================================
// k2: fused scores + softmax + PV.  grid = (S/16, H, B), 8 waves.
//   LDS: P[16][2064] bf16 (66048 B) + red[8*16] f32 (512 B), dynamic.
__global__ __launch_bounds__(256) void attn_kernel(
    const unsigned short* __restrict__ Qb,
    const unsigned short* __restrict__ Kb,
    const unsigned short* __restrict__ Vt,
    const float* __restrict__ amask,
    unsigned short* __restrict__ AOb,
    float* __restrict__ attnw) {
    extern __shared__ char smem[];
    unsigned short (*P)[2064] = (unsigned short (*)[2064])smem;
    float* red = (float*)(smem + (size_t)16 * 2064 * 2);

    const int h     = blockIdx.y;
    const int bi    = blockIdx.z;
    const int tid   = threadIdx.x;
    const int wave  = tid >> 5;
    const int lane  = tid & 31;
    const int lhalf = lane & 15;
    const int lhi   = lane >> 4;
    const int s0    = blockIdx.x * 16;
    const int nbase = wave * 256;       // this wave's key range [nbase, nbase+256)

    // Q fragments for all of K-dim (8 x 16x32 bf16)
    const unsigned short* qrow =
        Qb + (((size_t)bi * HHN + h) * SSQ + s0 + lhalf) * DDH;
    Frag qf[8];
#pragma unroll
    for (int kk = 0; kk < 8; ++kk) {
        qf[kk].u[0] = *(const U4*)(qrow + kk * 32 + 16 * lhi);
        qf[kk].u[1] = *(const U4*)(qrow + kk * 32 + 16 * lhi + 8);
    }

    // --- scores: 16 key tiles of 16 per wave -------------------------------
    f32x8 lg[16];
#pragma unroll
    for (int t = 0; t < 16; ++t) {
        f32x8 acc = {};
        const unsigned short* krow =
            Kb + ((size_t)bi * SSQ + nbase + t * 16 + lhalf) * DDH;
#pragma unroll
        for (int kk = 0; kk < 8; ++kk) {
            Frag bf;
            bf.u[0] = *(const U4*)(krow + kk * 32 + 16 * lhi);
            bf.u[1] = *(const U4*)(krow + kk * 32 + 16 * lhi + 8);
            acc = wmma_bf16(qf[kk].v, bf.v, acc);
        }
#pragma unroll
        for (int r = 0; r < 8; ++r) {
            int qr   = s0 + r + 8 * lhi;
            int ncol = nbase + t * 16 + lhalf;
            float m  = amask[((size_t)bi * SSQ + qr) * SSQ + ncol];
            lg[t][r] = acc[r] * 0.0625f + m;     // 1/sqrt(256) = 1/16
        }
    }

    // --- softmax over full 2048 keys (shuffle + LDS cross-wave) ------------
    float mx[8];
#pragma unroll
    for (int r = 0; r < 8; ++r) {
        float v = lg[0][r];
#pragma unroll
        for (int t = 1; t < 16; ++t) v = fmaxf(v, lg[t][r]);
#pragma unroll
        for (int off = 1; off < 16; off <<= 1) v = fmaxf(v, __shfl_xor(v, off, 32));
        mx[r] = v;
    }
    if (lhalf == 0) {
#pragma unroll
        for (int r = 0; r < 8; ++r) red[wave * 16 + r + 8 * lhi] = mx[r];
    }
    __syncthreads();
    float gmx[8];
#pragma unroll
    for (int r = 0; r < 8; ++r) {
        float v = -3.0e38f;
#pragma unroll
        for (int w2 = 0; w2 < 8; ++w2) v = fmaxf(v, red[w2 * 16 + r + 8 * lhi]);
        gmx[r] = v;
    }
    __syncthreads();
    float sm[8];
#pragma unroll
    for (int r = 0; r < 8; ++r) {
        float s = 0.f;
#pragma unroll
        for (int t = 0; t < 16; ++t) {
            float e = __expf(lg[t][r] - gmx[r]);
            lg[t][r] = e;
            s += e;
        }
#pragma unroll
        for (int off = 1; off < 16; off <<= 1) s += __shfl_xor(s, off, 32);
        sm[r] = s;
    }
    if (lhalf == 0) {
#pragma unroll
        for (int r = 0; r < 8; ++r) red[wave * 16 + r + 8 * lhi] = sm[r];
    }
    __syncthreads();
    float pinv[8];
#pragma unroll
    for (int r = 0; r < 8; ++r) {
        float s = 0.f;
#pragma unroll
        for (int w2 = 0; w2 < 8; ++w2) s += red[w2 * 16 + r + 8 * lhi];
        pinv[r] = 1.0f / s;
    }

    // --- emit attn_weights (f32, non-temporal) and P (bf16 in LDS) ---------
    float* aw = attnw + (((size_t)bi * HHN + h) * SSQ + s0) * SSQ;
#pragma unroll
    for (int t = 0; t < 16; ++t) {
#pragma unroll
        for (int r = 0; r < 8; ++r) {
            int rl   = r + 8 * lhi;
            int ncol = nbase + t * 16 + lhalf;
            float p  = lg[t][r] * pinv[r];
            __builtin_nontemporal_store(p, &aw[(size_t)rl * SSQ + ncol]);
            P[rl][ncol] = f2bf(p);
        }
    }
    __syncthreads();

    // --- PV: out(16x256) = P(16x2048) @ V(2048x256), V transposed ----------
    const int n0a = 2 * wave * 16;
    const int n0b = n0a + 16;
    const unsigned short* v0 = Vt + ((size_t)bi * DDH + n0a + lhalf) * SSQ;
    const unsigned short* v1 = Vt + ((size_t)bi * DDH + n0b + lhalf) * SSQ;
    const unsigned short* prow = P[lhalf];
    f32x8 c0 = {}, c1 = {};
    for (int k0 = 0; k0 < SSQ; k0 += 32) {
        Frag a, b0, b1;
        a.u[0]  = *(const U4*)(prow + k0 + 16 * lhi);
        a.u[1]  = *(const U4*)(prow + k0 + 16 * lhi + 8);
        b0.u[0] = *(const U4*)(v0 + k0 + 16 * lhi);
        b0.u[1] = *(const U4*)(v0 + k0 + 16 * lhi + 8);
        b1.u[0] = *(const U4*)(v1 + k0 + 16 * lhi);
        b1.u[1] = *(const U4*)(v1 + k0 + 16 * lhi + 8);
        c0 = wmma_bf16(a.v, b0.v, c0);
        c1 = wmma_bf16(a.v, b1.v, c1);
    }
#pragma unroll
    for (int r = 0; r < 8; ++r) {
        int srow = s0 + r + 8 * lhi;
        size_t base = ((size_t)bi * SSQ + srow) * HIDN + (size_t)h * DDH;
        AOb[base + n0a + lhalf] = f2bf(c0[r]);
        AOb[base + n0b + lhalf] = f2bf(c1[r]);
    }
}

// ===========================================================================
// k3: output projection AOb(4096x2048) @ Wo -> f32.
//   grid = (BSR/32, 8). WG tile 32x256; wave w -> cols [32w, 32w+32), both
//   16-row halves (2Mx2N blocking). A-chunks (32x64 bf16) staged into LDS by
//   the Tensor Data Mover (double buffered); LDS rows padded to 144 B so the
//   16 rows a half-wave reads land in 16 distinct banks.
__global__ __launch_bounds__(256) void oproj_kernel(
    const unsigned short* __restrict__ AOb,
    const unsigned short* __restrict__ Wot,
    float* __restrict__ out) {
    __shared__ unsigned short Abuf[2][ACH_ROWS * ACH_STRH];

    const int tid   = threadIdx.x;
    const int wave  = tid >> 5;
    const int lane  = tid & 31;
    const int lhalf = lane & 15;
    const int lhi   = lane >> 4;
    const int m0    = blockIdx.x * 32;
    const int n0    = blockIdx.y * 256 + wave * 32;

    const unsigned short* brow0 = Wot + (size_t)(n0 + lhalf) * HIDN;
    const unsigned short* brow1 = Wot + (size_t)(n0 + 16 + lhalf) * HIDN;

    f32x8 c00 = {}, c01 = {}, c10 = {}, c11 = {};

#ifdef HAVE_TDM
    const unsigned lds_base = (unsigned)(size_t)(&Abuf[0][0]);
    const unsigned long long gbase =
        (unsigned long long)(size_t)(AOb + (size_t)m0 * HIDN);
    if (wave == 0) {
        tdm_load_2d(lds_base, gbase, ACH_K, ACH_ROWS, HIDN, BSR, HIDN, TDM_CFG);
        tdm_load_2d(lds_base + ACH_BYTES, gbase + 2 * ACH_K,
                    ACH_K, ACH_ROWS, HIDN, BSR, HIDN, TDM_CFG);
        __builtin_amdgcn_s_wait_tensorcnt(1);      // chunk 0 complete
    }
    __syncthreads();
    for (int c = 0; c < NCHUNK; ++c) {
        const unsigned short* abuf = &Abuf[c & 1][0];
        const int kg = c * ACH_K;
#pragma unroll
        for (int ks = 0; ks < 2; ++ks) {
            const int kc = ks * 32;
            Frag a0, a1, b0, b1;
            const unsigned short* ar0 = abuf + (size_t)lhalf * ACH_STRH;
            const unsigned short* ar1 = abuf + (size_t)(16 + lhalf) * ACH_STRH;
            a0.u[0] = *(const U4*)(ar0 + kc + 16 * lhi);
            a0.u[1] = *(const U4*)(ar0 + kc + 16 * lhi + 8);
            a1.u[0] = *(const U4*)(ar1 + kc + 16 * lhi);
            a1.u[1] = *(const U4*)(ar1 + kc + 16 * lhi + 8);
            const int k0 = kg + kc;
            b0.u[0] = *(const U4*)(brow0 + k0 + 16 * lhi);
            b0.u[1] = *(const U4*)(brow0 + k0 + 16 * lhi + 8);
            b1.u[0] = *(const U4*)(brow1 + k0 + 16 * lhi);
            b1.u[1] = *(const U4*)(brow1 + k0 + 16 * lhi + 8);
            c00 = wmma_bf16(a0.v, b0.v, c00);
            c01 = wmma_bf16(a0.v, b1.v, c01);
            c10 = wmma_bf16(a1.v, b0.v, c10);
            c11 = wmma_bf16(a1.v, b1.v, c11);
        }
        __syncthreads();                    // all waves done reading buf[c&1]
        if (wave == 0) {
            if (c + 2 < NCHUNK) {
                tdm_load_2d(lds_base + (unsigned)((c & 1) * ACH_BYTES),
                            gbase + (unsigned long long)(c + 2) * (2 * ACH_K),
                            ACH_K, ACH_ROWS, HIDN, BSR, HIDN, TDM_CFG);
                __builtin_amdgcn_s_wait_tensorcnt(1);   // chunk c+1 complete
            } else {
                __builtin_amdgcn_s_wait_tensorcnt(0);   // drain
            }
        }
        __syncthreads();                    // publish chunk c+1 to all waves
    }
#else
    for (int c = 0; c < NCHUNK; ++c) {
        const int kg = c * ACH_K;
        __syncthreads();
        for (int i = tid; i < ACH_ROWS * ACH_K; i += 256) {
            int r = i >> 6, kk = i & 63;
            Abuf[0][r * ACH_STRH + kk] = AOb[(size_t)(m0 + r) * HIDN + kg + kk];
        }
        __syncthreads();
        const unsigned short* abuf = &Abuf[0][0];
#pragma unroll
        for (int ks = 0; ks < 2; ++ks) {
            const int kc = ks * 32;
            Frag a0, a1, b0, b1;
            const unsigned short* ar0 = abuf + (size_t)lhalf * ACH_STRH;
            const unsigned short* ar1 = abuf + (size_t)(16 + lhalf) * ACH_STRH;
            a0.u[0] = *(const U4*)(ar0 + kc + 16 * lhi);
            a0.u[1] = *(const U4*)(ar0 + kc + 16 * lhi + 8);
            a1.u[0] = *(const U4*)(ar1 + kc + 16 * lhi);
            a1.u[1] = *(const U4*)(ar1 + kc + 16 * lhi + 8);
            const int k0 = kg + kc;
            b0.u[0] = *(const U4*)(brow0 + k0 + 16 * lhi);
            b0.u[1] = *(const U4*)(brow0 + k0 + 16 * lhi + 8);
            b1.u[0] = *(const U4*)(brow1 + k0 + 16 * lhi);
            b1.u[1] = *(const U4*)(brow1 + k0 + 16 * lhi + 8);
            c00 = wmma_bf16(a0.v, b0.v, c00);
            c01 = wmma_bf16(a0.v, b1.v, c01);
            c10 = wmma_bf16(a1.v, b0.v, c10);
            c11 = wmma_bf16(a1.v, b1.v, c11);
        }
    }
#endif

#pragma unroll
    for (int r = 0; r < 8; ++r) {
        int rr = r + 8 * lhi;
        out[(size_t)(m0 + rr) * HIDN + n0 + lhalf]           = c00[r];
        out[(size_t)(m0 + rr) * HIDN + n0 + 16 + lhalf]      = c01[r];
        out[(size_t)(m0 + 16 + rr) * HIDN + n0 + lhalf]      = c10[r];
        out[(size_t)(m0 + 16 + rr) * HIDN + n0 + 16 + lhalf] = c11[r];
    }
}

// ===========================================================================
extern "C" void kernel_launch(void* const* d_in, const int* in_sizes, int n_in,
                              void* d_out, int out_size, void* d_ws, size_t ws_size,
                              hipStream_t stream) {
    (void)in_sizes; (void)n_in; (void)out_size; (void)ws_size;

    const float* X     = (const float*)d_in[0];
    const float* amask = (const float*)d_in[1];
    const int*   pid   = (const int*)d_in[2];
    const float* Wq    = (const float*)d_in[3];
    const float* Wk    = (const float*)d_in[4];
    const float* Wv    = (const float*)d_in[5];
    const float* Wo    = (const float*)d_in[6];

    float* out   = (float*)d_out;
    float* attnw = out + (size_t)BSR * HIDN;     // after attn_output

    char* ws = (char*)d_ws;
    unsigned short* Xb  = (unsigned short*)(ws + OFF_XB);
    unsigned short* Wqt = (unsigned short*)(ws + OFF_WQT);
    unsigned short* Wkt = (unsigned short*)(ws + OFF_WKT);
    unsigned short* Wvt = (unsigned short*)(ws + OFF_WVT);
    unsigned short* Wot = (unsigned short*)(ws + OFF_WOT);
    unsigned short* Qb  = (unsigned short*)(ws + OFF_QB);
    unsigned short* Kb  = (unsigned short*)(ws + OFF_KB);
    unsigned short* Vt  = (unsigned short*)(ws + OFF_VT);
    unsigned short* AOb = (unsigned short*)(ws + OFF_AOB);

    // k0: convert / transpose operands to bf16
    cvt_kernel<<<(BSR * HIDN + 255) / 256, 256, 0, stream>>>(X, Xb, BSR * HIDN);
    cvt_t_kernel<<<(HIDN * HIDN + 255) / 256, 256, 0, stream>>>(Wq, Wqt, HIDN, HIDN);
    cvt_t_kernel<<<(HIDN * DDH + 255) / 256, 256, 0, stream>>>(Wk, Wkt, HIDN, DDH);
    cvt_t_kernel<<<(HIDN * DDH + 255) / 256, 256, 0, stream>>>(Wv, Wvt, HIDN, DDH);
    cvt_t_kernel<<<(HIDN * HIDN + 255) / 256, 256, 0, stream>>>(Wo, Wot, HIDN, HIDN);

    // k1: QKV projection + RoPE (y: 8 Q heads, K, V), TDM-staged A
    qkv_kernel<<<dim3(BSR / 32, 10), 256, 0, stream>>>(Xb, Wqt, Wkt, Wvt, pid,
                                                       Qb, Kb, Vt);

    // k2: fused attention (scores + softmax + PV), 66.5 KB dynamic LDS
    size_t smem = (size_t)16 * 2064 * 2 + (size_t)8 * 16 * 4;
    attn_kernel<<<dim3(SSQ / 16, HHN, BB), 256, smem, stream>>>(Qb, Kb, Vt, amask,
                                                                AOb, attnw);

    // k3: output projection (TDM-staged A, 32x256 tiles)
    oproj_kernel<<<dim3(BSR / 32, HIDN / 256), 256, 0, stream>>>(AOb, Wot, out);
}
